// PHMLinear_78280073937084
// MI455X (gfx1250) — matrix operand order repaired and checked by reading, not compiled, not run on hardware
//
#include <hip/hip_runtime.h>
#include <hip/hip_bf16.h>

// PHM linear: y = x @ W^T + bias, W = sum_b A[b] (kron) B[b]
// M=16384, K=4096, N=4096, RANK=8.
//
// MI455X plan:
//  1) x fp32 -> bf16 workspace (128 MB)
//  2) W materialized in bf16 (32 MB)  -> both L2-resident (160 MB < 192 MB)
//  3) WMMA GEMM: 128x128 block tile (4 waves, 64x64 each), K-step 64.
//     Each K-step's A/B tiles are DMAed to LDS by the Tensor Data Mover as
//     two 128x32 descriptors (keeps the bank-conflict-free 80 B row stride),
//     double-buffered and TENSORcnt-synchronized; fragments via ds_load_b128.

typedef __attribute__((ext_vector_type(16))) __bf16 v16bf;
typedef __attribute__((ext_vector_type(8)))  __bf16 v8bf;
typedef __attribute__((ext_vector_type(8)))  float  v8f;
typedef __attribute__((ext_vector_type(4)))  unsigned int u32x4;
typedef __attribute__((ext_vector_type(8)))  int i32x8;
typedef __attribute__((ext_vector_type(4)))  int i32x4;

typedef __attribute__((address_space(3))) __bf16 lds_bf16;
typedef __attribute__((address_space(3))) v8bf  lds_v8bf;

#define M_DIM   16384
#define K_DIM   4096
#define N_DIM   4096
#define RANK    8

// One 32-K chunk in LDS: 128 rows x 32 bf16 (64 B) + 16 B pad -> 80 B stride.
#define LDS_ROW   40
#define TILE_ELEM (128 * LDS_ROW)   // 5120 elements = 10240 B per chunk

// ---------------------------------------------------------------------------
// Kernel 1: x fp32 -> bf16
// ---------------------------------------------------------------------------
__global__ __launch_bounds__(256) void cvt_x_bf16(const float* __restrict__ X,
                                                  __bf16* __restrict__ Xb) {
    size_t idx = ((size_t)blockIdx.x * blockDim.x + threadIdx.x) * 8;
    const float4 f0 = *(const float4*)(X + idx);
    const float4 f1 = *(const float4*)(X + idx + 4);
    v8bf o;
    o[0] = (__bf16)f0.x; o[1] = (__bf16)f0.y; o[2] = (__bf16)f0.z; o[3] = (__bf16)f0.w;
    o[4] = (__bf16)f1.x; o[5] = (__bf16)f1.y; o[6] = (__bf16)f1.z; o[7] = (__bf16)f1.w;
    *(v8bf*)(Xb + idx) = o;
}

// ---------------------------------------------------------------------------
// Kernel 2: W[o][c] = sum_b A[b, o>>6, c>>6] * B[b, o&63, c&63]  (bf16 out)
// ---------------------------------------------------------------------------
__global__ __launch_bounds__(256) void build_w_bf16(const float* __restrict__ A,
                                                    const float* __restrict__ B,
                                                    __bf16* __restrict__ Wb) {
    int tid = blockIdx.x * blockDim.x + threadIdx.x;
    int o   = tid >> 9;
    int c0  = (tid & 511) * 8;
    int i = o >> 6, k = o & 63;
    int j = c0 >> 6, m0 = c0 & 63;

    float s[8] = {0.f, 0.f, 0.f, 0.f, 0.f, 0.f, 0.f, 0.f};
#pragma unroll
    for (int b = 0; b < RANK; ++b) {
        const float a = A[b * 4096 + i * 64 + j];
        const float4 b0 = *(const float4*)(B + b * 4096 + k * 64 + m0);
        const float4 b1 = *(const float4*)(B + b * 4096 + k * 64 + m0 + 4);
        s[0] += a * b0.x; s[1] += a * b0.y; s[2] += a * b0.z; s[3] += a * b0.w;
        s[4] += a * b1.x; s[5] += a * b1.y; s[6] += a * b1.z; s[7] += a * b1.w;
    }
    v8bf out;
#pragma unroll
    for (int e = 0; e < 8; ++e) out[e] = (__bf16)s[e];
    *(v8bf*)(Wb + (size_t)o * K_DIM + c0) = out;
}

// ---------------------------------------------------------------------------
// TDM: DMA one 128-row x 32-col bf16 tile (row stride K_DIM elems in memory)
// into LDS with 16 B padding after every 64 B row (LDS stride 80 B).
// D# layout per CDNA5 ISA ch.8; 6-arg builtin on this toolchain.
// ---------------------------------------------------------------------------
static __device__ __forceinline__ void tdm_load_tile(const __bf16* gtile,
                                                     unsigned lds_byte_addr,
                                                     unsigned tensor_rows) {
    unsigned long long ga = (unsigned long long)(const void*)gtile;
    u32x4 g0;
    g0[0] = 1u;                                      // count=1, user descriptor
    g0[1] = lds_byte_addr;                           // lds_addr
    g0[2] = (unsigned)ga;                            // global_addr[31:0]
    g0[3] = (unsigned)((ga >> 32) & 0x01FFFFFFu)     // global_addr[56:32]
          | (2u << 30);                              // type=2 ("image")
    i32x8 g1;
    g1[0] = (int)((1u << 16)                         // data_size=1 -> 2 bytes
                | (1u << 20)                         // pad_enable
                | (3u << 22)                         // pad_interval: 16 DWORDs (64 B)
                | (3u << 25));                       // pad_amount: 4 DWORDs (16 B)
    g1[1] = (int)((unsigned)(K_DIM & 0xFFFF) << 16); // tensor_dim0 = 4096 (lo16)
    g1[2] = (int)((tensor_rows & 0xFFFFu) << 16);    // dim0 hi=0 | tensor_dim1 lo16
    g1[3] = (int)((32u << 16) | (tensor_rows >> 16));// tensor_dim1 hi | tile_dim0=32
    g1[4] = 128;                                     // tile_dim1 = 128 rows (tile_dim2=0)
    g1[5] = K_DIM;                                   // tensor_dim0_stride lo32
    g1[6] = 0;
    g1[7] = 0;
    i32x4 z4 = {0, 0, 0, 0};
    i32x8 z8 = {0, 0, 0, 0, 0, 0, 0, 0};
    __builtin_amdgcn_tensor_load_to_lds(g0, g1, z4, z4, z8, 0);
}

// ---------------------------------------------------------------------------
// Kernel 3: WMMA GEMM, TDM double-buffered LDS staging, K-step 64.
// ---------------------------------------------------------------------------
__global__ __launch_bounds__(128) void gemm_wmma_bf16(const __bf16* __restrict__ Xb,
                                                      const __bf16* __restrict__ Wb,
                                                      const float* __restrict__ bias,
                                                      float* __restrict__ Y) {
    // [buf][A=0/B=1][kchunk][chunk elems] = 80 KB
    __shared__ __align__(128) __bf16 smem[2][2][2][TILE_ELEM];

    const int lane = threadIdx.x & 31;
    const int wave = threadIdx.x >> 5;
    const int lr = lane & 15;
    const int hi = lane >> 4;

    const int mblk = blockIdx.x * 128;
    const int nblk = blockIdx.y * 128;
    const int mw = (wave >> 1) * 64;   // wave offsets inside the block tile
    const int nw = (wave & 1) * 64;

    // LDS byte addresses for the TDM descriptors.
    unsigned lds_addr[2][2][2];
#pragma unroll
    for (int b = 0; b < 2; ++b)
#pragma unroll
        for (int w = 0; w < 2; ++w)
#pragma unroll
            for (int c = 0; c < 2; ++c)
                lds_addr[b][w][c] =
                    (unsigned)(unsigned long long)(const void*)&smem[b][w][c][0];

    // Laundered LDS pointer: the TDM writes are invisible to alias analysis.
    const __bf16* smg = &smem[0][0][0][0];
    asm volatile("" : "+v"(smg) :: "memory");
    const lds_bf16* sm = (const lds_bf16*)smg;

    v8f acc[4][4];
#pragma unroll
    for (int t = 0; t < 4; ++t)
#pragma unroll
        for (int u = 0; u < 4; ++u) acc[t][u] = (v8f){0.f,0.f,0.f,0.f,0.f,0.f,0.f,0.f};

    const __bf16* Abase = Xb + (size_t)mblk * K_DIM;
    const __bf16* Bbase = Wb + (size_t)nblk * K_DIM;

    // Prologue: DMA first 64-K step into buffer 0 (wave0: A, wave1: B).
    if (wave == 0) {
        tdm_load_tile(Abase,      lds_addr[0][0][0], M_DIM);
        tdm_load_tile(Abase + 32, lds_addr[0][0][1], M_DIM);
    }
    if (wave == 1) {
        tdm_load_tile(Bbase,      lds_addr[0][1][0], N_DIM);
        tdm_load_tile(Bbase + 32, lds_addr[0][1][1], N_DIM);
    }

    const int KSTEPS = K_DIM / 64;
    for (int i = 0; i < KSTEPS; ++i) {
        const int cur = i & 1;

        if (wave < 2) __builtin_amdgcn_s_wait_tensorcnt(0); // buf[cur] pair done
        __syncthreads();   // buf[cur] visible to all; all waves are past their
                           // reads of buf[1-cur] (finished previous iteration)
        if (i + 1 < KSTEPS) {
            const int k1 = (i + 1) * 64;
            const int nxt = 1 - cur;
            if (wave == 0) {
                tdm_load_tile(Abase + k1,      lds_addr[nxt][0][0], M_DIM);
                tdm_load_tile(Abase + k1 + 32, lds_addr[nxt][0][1], M_DIM);
            }
            if (wave == 1) {
                tdm_load_tile(Bbase + k1,      lds_addr[nxt][1][0], N_DIM);
                tdm_load_tile(Bbase + k1 + 32, lds_addr[nxt][1][1], N_DIM);
            }
        }

        const lds_bf16* Ab = sm + ((cur * 2 + 0) * 2) * TILE_ELEM;
        const lds_bf16* Bb = sm + ((cur * 2 + 1) * 2) * TILE_ELEM;

        // Hoist B fragments for both k-chunks (shared across all 4 t-groups).
        v16bf bfrag[2][4];
#pragma unroll
        for (int c = 0; c < 2; ++c)
#pragma unroll
            for (int u = 0; u < 4; ++u) {
                const lds_bf16* pr =
                    Bb + c * TILE_ELEM + (nw + 16 * u + lr) * LDS_ROW + hi * 8;
                union { v16bf v; v8bf h[2]; } bu;
                bu.h[0] = *(const lds_v8bf*)(pr);
                bu.h[1] = *(const lds_v8bf*)(pr + 16);
                bfrag[c][u] = bu.v;
            }

#pragma unroll
        for (int t = 0; t < 4; ++t) {
            const int arow = (mw + 16 * t + lr) * LDS_ROW + hi * 8;
            union { v16bf v; v8bf h[2]; } a0, a1;
            {
                const lds_bf16* pr = Ab + arow;                // k-chunk 0
                a0.h[0] = *(const lds_v8bf*)(pr);
                a0.h[1] = *(const lds_v8bf*)(pr + 16);
            }
            {
                const lds_bf16* pr = Ab + TILE_ELEM + arow;    // k-chunk 1
                a1.h[0] = *(const lds_v8bf*)(pr);
                a1.h[1] = *(const lds_v8bf*)(pr + 16);
            }
#pragma unroll
            for (int u = 0; u < 4; ++u)
                acc[t][u] = __builtin_amdgcn_wmma_f32_16x16x32_bf16(
                    false, a0.v, false, bfrag[0][u], (short)0, acc[t][u], false, false);
#pragma unroll
            for (int u = 0; u < 4; ++u)
                acc[t][u] = __builtin_amdgcn_wmma_f32_16x16x32_bf16(
                    false, a1.v, false, bfrag[1][u], (short)0, acc[t][u], false, false);
        }
    }

    // Epilogue: C/D layout -> element (M = r + 8*hi, N = lr) is acc[r].
#pragma unroll
    for (int u = 0; u < 4; ++u) {
        const int col = nblk + nw + 16 * u + lr;
        const float bv = bias[col];
#pragma unroll
        for (int t = 0; t < 4; ++t) {
            const int rowbase = mblk + mw + 16 * t + hi * 8;
            float* yp = Y + (size_t)rowbase * N_DIM + col;
#pragma unroll
            for (int r = 0; r < 8; ++r)
                yp[(size_t)r * N_DIM] = acc[t][u][r] + bv;
        }
    }
}

// ---------------------------------------------------------------------------
extern "C" void kernel_launch(void* const* d_in, const int* in_sizes, int n_in,
                              void* d_out, int out_size, void* d_ws, size_t ws_size,
                              hipStream_t stream) {
    const float* x    = (const float*)d_in[0];  // (8,2048,4096)
    const float* A    = (const float*)d_in[1];  // (8,64,64)
    const float* B    = (const float*)d_in[2];  // (8,64,64)
    const float* bias = (const float*)d_in[3];  // (4096,)
    float* y = (float*)d_out;

    // Workspace: [ x_bf16 : 128 MB ][ W_bf16 : 32 MB ]
    __bf16* xb = (__bf16*)d_ws;
    __bf16* wb = (__bf16*)((char*)d_ws + (size_t)M_DIM * K_DIM * sizeof(__bf16));

    cvt_x_bf16<<<(M_DIM * (size_t)K_DIM) / 8 / 256, 256, 0, stream>>>(x, xb);
    build_w_bf16<<<(N_DIM * (size_t)K_DIM) / 8 / 256, 256, 0, stream>>>(A, B, wb);

    dim3 grid(M_DIM / 128, N_DIM / 128);
    gemm_wmma_bf16<<<grid, 128, 0, stream>>>(xb, wb, bias, y);
}
